// HHGTLayer_30408368456301
// MI455X (gfx1250) — compile-verified, matrix operand order checked by reading, not run on hardware
//
#include <hip/hip_runtime.h>

// Problem constants (match reference setup_inputs)
#define NN 40000      // nodes per type
#define EE 120000     // incidences per type
#define HEN 16384     // hyperedges
#define CC 512        // channels
#define HH 8          // heads
// DK = 64, scale = 1/sqrt(64) = 0.125

typedef __attribute__((ext_vector_type(16))) __bf16 v16bf;
typedef __attribute__((ext_vector_type(8)))  __bf16 v8bf;
typedef __attribute__((ext_vector_type(8)))  float  v8f;

__device__ __forceinline__ unsigned short f32_bf16(float f) {
  unsigned int u = __float_as_uint(f);
  u += 0x7FFFu + ((u >> 16) & 1u);   // round-to-nearest-even
  return (unsigned short)(u >> 16);
}

// ---------------------------------------------------------------- utilities
__global__ __launch_bounds__(256) void k_cvt_bf16(const float* __restrict__ in,
                                                  unsigned short* __restrict__ out, int n4) {
  int t = blockIdx.x * 256 + threadIdx.x;
  if (t >= n4) return;
  const float4 v = reinterpret_cast<const float4*>(in)[t];
  uint2 p;
  p.x = (unsigned int)f32_bf16(v.x) | ((unsigned int)f32_bf16(v.y) << 16);
  p.y = (unsigned int)f32_bf16(v.z) | ((unsigned int)f32_bf16(v.w) << 16);
  reinterpret_cast<uint2*>(out)[t] = p;
}

// Wt[n][k] = bf16(W[k][n]) : 512x512 transpose+convert (run once per weight)
__global__ __launch_bounds__(256) void k_cvt_bf16_T(const float* __restrict__ W,
                                                    unsigned short* __restrict__ Wt) {
  int t = blockIdx.x * 256 + threadIdx.x;      // over CC*CC/4 outputs
  if (t >= (CC * CC) / 4) return;
  int n  = t >> 7;                             // 0..511
  int k4 = (t & 127) << 2;                     // 0..508
  uint2 p;
  p.x = (unsigned int)f32_bf16(W[(size_t)(k4 + 0) * CC + n]) |
        ((unsigned int)f32_bf16(W[(size_t)(k4 + 1) * CC + n]) << 16);
  p.y = (unsigned int)f32_bf16(W[(size_t)(k4 + 2) * CC + n]) |
        ((unsigned int)f32_bf16(W[(size_t)(k4 + 3) * CC + n]) << 16);
  reinterpret_cast<uint2*>(Wt)[t] = p;
}

__global__ __launch_bounds__(256) void k_zero4(float* __restrict__ p, int n4) {
  int t = blockIdx.x * 256 + threadIdx.x;
  if (t >= n4) return;
  float4 z; z.x = z.y = z.z = z.w = 0.f;
  reinterpret_cast<float4*>(p)[t] = z;
}

__global__ void k_init_stats(float* __restrict__ stats) {
  int t = threadIdx.x;
  if (t < 16) stats[t] = -3.4e38f;      // per-head maxima (a:0-7, b:8-15)
  else if (t < 32) stats[t] = 0.f;      // per-head sums   (a:16-23, b:24-31)
}

// ------------------------------------------------------- WMMA bf16 GEMM
// C[M,512] = A[M,512](bf16, row-major) @ Bt[512,512](bf16, PRE-TRANSPOSED [n][k]) + bias
// Block: 256 threads = 8 waves; tile 128(M) x 256(N); each wave owns 64x64
// (4x4 WMMA tiles) -> 16 v_wmma per k-step of 32, 16 ds_load_b128 per k-step.
__global__ __launch_bounds__(256) void k_gemm_bf16(
    const unsigned short* __restrict__ A, const unsigned short* __restrict__ Bt,
    const float* __restrict__ bias, float* __restrict__ C, int M) {
  __shared__ unsigned short Alds[128 * 32];      // [m][k]
  __shared__ unsigned short Blds[256 * 32];      // [n][k]
  const int tid  = threadIdx.x;
  const int lane = tid & 31;
  const int wave = tid >> 5;
  const int wm   = wave >> 2;                    // 0..1 -> M strip of 64
  const int wn   = wave & 3;                     // 0..3 -> N strip of 64
  const int hf   = lane >> 4;                    // K-half select (wave32 layout)
  const int l16  = lane & 15;
  const int row0 = blockIdx.y * 128;
  const int col0 = blockIdx.x * 256;

  v8f acc[4][4];
#pragma unroll
  for (int mi = 0; mi < 4; ++mi)
#pragma unroll
    for (int ni = 0; ni < 4; ++ni)
#pragma unroll
      for (int i = 0; i < 8; ++i) acc[mi][ni][i] = 0.f;

  for (int k0 = 0; k0 < CC; k0 += 32) {
    // ---- stage A tile (128x32): thread -> 16 contiguous bf16 (2x b128)
    {
      int r  = tid >> 1;
      int kc = (tid & 1) * 16;
      int4 z0, z1;
      z0.x = z0.y = z0.z = z0.w = 0; z1 = z0;
      if (row0 + r < M) {
        const int4* g = reinterpret_cast<const int4*>(A + (size_t)(row0 + r) * CC + k0 + kc);
        z0 = g[0]; z1 = g[1];
        if (k0 + 32 < CC)   // prefetch next k-tile (global_prefetch_b8)
          __builtin_prefetch(A + (size_t)(row0 + r) * CC + k0 + 32 + kc, 0, 0);
      }
      int4* s = reinterpret_cast<int4*>(&Alds[r * 32 + kc]);
      s[0] = z0; s[1] = z1;
    }
    // ---- stage B tile (256 n x 32 k) from pre-transposed Bt: pure b128 copies
    {
      const int4* g = reinterpret_cast<const int4*>(Bt + (size_t)(col0 + tid) * CC + k0);
      int4* s = reinterpret_cast<int4*>(&Blds[tid * 32]);
      int4 b0 = g[0], b1 = g[1], b2 = g[2], b3 = g[3];
      s[0] = b0; s[1] = b1; s[2] = b2; s[3] = b3;
    }
    __syncthreads();

    const __bf16* Ab = reinterpret_cast<const __bf16*>(Alds);
    const __bf16* Bb = reinterpret_cast<const __bf16*>(Blds);

    // A fragments: lane l16 = row; K runs [hf*8, hf*8+8) and [16+hf*8, ...)
    v16bf afr[4];
#pragma unroll
    for (int mi = 0; mi < 4; ++mi) {
      int r = wm * 64 + mi * 16 + l16;
      v8bf lo = *reinterpret_cast<const v8bf*>(&Ab[r * 32 + hf * 8]);
      v8bf hi = *reinterpret_cast<const v8bf*>(&Ab[r * 32 + 16 + hf * 8]);
#pragma unroll
      for (int j = 0; j < 8; ++j) { afr[mi][j] = lo[j]; afr[mi][j + 8] = hi[j]; }
    }
    // B fragments: lane l16 = column
    v16bf bfr[4];
#pragma unroll
    for (int ni = 0; ni < 4; ++ni) {
      int n = wn * 64 + ni * 16 + l16;
      v8bf lo = *reinterpret_cast<const v8bf*>(&Bb[n * 32 + hf * 8]);
      v8bf hi = *reinterpret_cast<const v8bf*>(&Bb[n * 32 + 16 + hf * 8]);
#pragma unroll
      for (int j = 0; j < 8; ++j) { bfr[ni][j] = lo[j]; bfr[ni][j + 8] = hi[j]; }
    }
#pragma unroll
    for (int mi = 0; mi < 4; ++mi)
#pragma unroll
      for (int ni = 0; ni < 4; ++ni)
        acc[mi][ni] = __builtin_amdgcn_wmma_f32_16x16x32_bf16(
            false, afr[mi], false, bfr[ni], (short)0, acc[mi][ni], false, false);
    __syncthreads();
  }

  // epilogue: C/D layout -> row = i + hf*8, col = l16 within each 16x16 tile
#pragma unroll
  for (int mi = 0; mi < 4; ++mi)
#pragma unroll
    for (int ni = 0; ni < 4; ++ni) {
      int c = col0 + wn * 64 + ni * 16 + l16;
      float bb = bias[c];
#pragma unroll
      for (int i = 0; i < 8; ++i) {
        int r = row0 + wm * 64 + mi * 16 + hf * 8 + i;
        if (r < M) C[(size_t)r * CC + c] = acc[mi][ni][i] + bb;
      }
    }
}

// --------------------------------------------------- hypergraph scatter/softmax
__global__ __launch_bounds__(256) void k_scatter_he(
    const float* __restrict__ K, const int* __restrict__ nidx,
    const int* __restrict__ hidx, float* __restrict__ he) {
  int t = blockIdx.x * 256 + threadIdx.x;
  if (t >= EE * 128) return;
  int e = t >> 7, c = (t & 127) << 2;
  const float4 v = *reinterpret_cast<const float4*>(K + (size_t)nidx[e] * CC + c);
  float* d = he + (size_t)hidx[e] * CC + c;
  atomicAdd(d + 0, v.x); atomicAdd(d + 1, v.y);
  atomicAdd(d + 2, v.z); atomicAdd(d + 3, v.w);
}

__device__ __forceinline__ void atomicMaxF(float* addr, float val) {
  unsigned int* ua = reinterpret_cast<unsigned int*>(addr);
  unsigned int old = *ua;
  while (__uint_as_float(old) < val) {
    unsigned int assumed = old;
    old = atomicCAS(ua, assumed, __float_as_uint(val));
    if (old == assumed) break;
  }
}

__global__ __launch_bounds__(256) void k_logits(
    const float* __restrict__ Q, const float* __restrict__ he,
    const int* __restrict__ nidx, const int* __restrict__ hidx,
    float* __restrict__ logit, float* __restrict__ hmax) {
  __shared__ float smax[HH];
  if (threadIdx.x < HH) smax[threadIdx.x] = -3.4e38f;
  __syncthreads();
  int t = blockIdx.x * 256 + threadIdx.x;
  if (t < EE * HH) {
    int e = t >> 3, h = t & 7;
    const float4* q = reinterpret_cast<const float4*>(Q + (size_t)nidx[e] * CC + h * 64);
    const float4* k = reinterpret_cast<const float4*>(he + (size_t)hidx[e] * CC + h * 64);
    float s = 0.f;
#pragma unroll
    for (int j = 0; j < 16; ++j) {
      float4 a = q[j], b = k[j];
      s += a.x * b.x + a.y * b.y + a.z * b.z + a.w * b.w;
    }
    s *= 0.125f;                    // 1/sqrt(DK)
    logit[t] = s;
    atomicMaxF(&smax[h], s);
  }
  __syncthreads();
  if (threadIdx.x < HH) atomicMaxF(&hmax[threadIdx.x], smax[threadIdx.x]);
}

__global__ __launch_bounds__(256) void k_expsum(
    float* __restrict__ logit, const float* __restrict__ hmax, float* __restrict__ hsum) {
  __shared__ float ssum[HH];
  if (threadIdx.x < HH) ssum[threadIdx.x] = 0.f;
  __syncthreads();
  int t = blockIdx.x * 256 + threadIdx.x;
  if (t < EE * HH) {
    int h = t & 7;
    float w = __expf(logit[t] - hmax[h]);
    logit[t] = w;
    atomicAdd(&ssum[h], w);
  }
  __syncthreads();
  if (threadIdx.x < HH) atomicAdd(&hsum[threadIdx.x], ssum[threadIdx.x]);
}

__global__ __launch_bounds__(256) void k_scatter_nodes(
    const float* __restrict__ he, const float* __restrict__ logit,
    const float* __restrict__ hsum, const int* __restrict__ nidx,
    const int* __restrict__ hidx, float* __restrict__ nodeout) {
  int t = blockIdx.x * 256 + threadIdx.x;
  if (t >= EE * 128) return;
  int e = t >> 7, c = (t & 127) << 2;
  int h = c >> 6;
  float w = logit[e * HH + h] / hsum[h];
  const float4 v = *reinterpret_cast<const float4*>(he + (size_t)hidx[e] * CC + c);
  float* d = nodeout + (size_t)nidx[e] * CC + c;
  atomicAdd(d + 0, v.x * w); atomicAdd(d + 1, v.y * w);
  atomicAdd(d + 2, v.z * w); atomicAdd(d + 3, v.w * w);
}

// --------------------------------------------------------------- layernorm
__global__ __launch_bounds__(128) void k_add_ln(
    const float* __restrict__ merged, const float* __restrict__ S,
    const float* __restrict__ gam, const float* __restrict__ bet,
    float* __restrict__ out) {
  int row = blockIdx.x;
  int tid = threadIdx.x;                 // 128 threads x 4 elems = 512
  int c = tid * 4;
  const float4 m = *reinterpret_cast<const float4*>(merged + (size_t)row * CC + c);
  const float4 s = *reinterpret_cast<const float4*>(S + (size_t)row * CC + c);
  float y0 = m.x + s.x, y1 = m.y + s.y, y2 = m.z + s.z, y3 = m.w + s.w;
  float sum = y0 + y1 + y2 + y3;
  float sq  = y0 * y0 + y1 * y1 + y2 * y2 + y3 * y3;
#pragma unroll
  for (int d = 16; d; d >>= 1) {
    sum += __shfl_xor(sum, d, 32);
    sq  += __shfl_xor(sq,  d, 32);
  }
  __shared__ float w1[4], w2[4];
  if ((tid & 31) == 0) { w1[tid >> 5] = sum; w2[tid >> 5] = sq; }
  __syncthreads();
  float ts = w1[0] + w1[1] + w1[2] + w1[3];
  float tq = w2[0] + w2[1] + w2[2] + w2[3];
  float mu  = ts * (1.f / CC);
  float var = tq * (1.f / CC) - mu * mu;
  float inv = rsqrtf(var + 1e-5f);
  float4 o;
  o.x = gam[c + 0] * (y0 - mu) * inv + bet[c + 0];
  o.y = gam[c + 1] * (y1 - mu) * inv + bet[c + 1];
  o.z = gam[c + 2] * (y2 - mu) * inv + bet[c + 2];
  o.w = gam[c + 3] * (y3 - mu) * inv + bet[c + 3];
  *reinterpret_cast<float4*>(out + (size_t)row * CC + c) = o;
}

// ------------------------------------------------------------------ launch
extern "C" void kernel_launch(void* const* d_in, const int* in_sizes, int n_in,
                              void* d_out, int out_size, void* d_ws, size_t ws_size,
                              hipStream_t stream) {
  (void)in_sizes; (void)n_in; (void)out_size; (void)ws_size;
  const float* x_a   = (const float*)d_in[0];
  const float* x_b   = (const float*)d_in[1];
  const int* nidx_a  = (const int*)d_in[2];
  const int* hidx_a  = (const int*)d_in[3];
  const int* nidx_b  = (const int*)d_in[4];
  const int* hidx_b  = (const int*)d_in[5];
  // d_in[6] = max_he_id (HEN fixed)
  const float* kW_a = (const float*)d_in[7],  *kb_a = (const float*)d_in[8];
  const float* qW_a = (const float*)d_in[9],  *qb_a = (const float*)d_in[10];
  const float* aW_a = (const float*)d_in[11], *ab_a = (const float*)d_in[12];
  const float* sW_a = (const float*)d_in[13], *sb_a = (const float*)d_in[14];
  const float* lng_a = (const float*)d_in[15], *lnb_a = (const float*)d_in[16];
  const float* kW_b = (const float*)d_in[17], *kb_b = (const float*)d_in[18];
  const float* qW_b = (const float*)d_in[19], *qb_b = (const float*)d_in[20];
  const float* aW_b = (const float*)d_in[21], *ab_b = (const float*)d_in[22];
  const float* sW_b = (const float*)d_in[23], *sb_b = (const float*)d_in[24];
  const float* lng_b = (const float*)d_in[25], *lnb_b = (const float*)d_in[26];

  // workspace carve
  char* ws = (char*)d_ws;
  size_t off = 0;
  auto carve = [&](size_t bytes) -> void* {
    void* p = ws + off;
    off = (off + bytes + 255) & ~(size_t)255;
    return p;
  };
  const size_t NCf = (size_t)NN * CC;
  unsigned short* xbf_a = (unsigned short*)carve(NCf * 2);
  unsigned short* xbf_b = (unsigned short*)carve(NCf * 2);
  unsigned short* wbf[8];                        // pre-transposed bf16 weights [n][k]
  for (int i = 0; i < 8; ++i) wbf[i] = (unsigned short*)carve((size_t)CC * CC * 2);
  // 0:kA 1:qA 2:sA 3:aA 4:kB 5:qB 6:sB 7:aB
  float* Ka = (float*)carve(NCf * 4);   // reused as node_out_a
  float* Kb = (float*)carve(NCf * 4);   // reused as node_out_b
  float* Qa = (float*)carve(NCf * 4);   // reused as merged_a
  float* Qb = (float*)carve(NCf * 4);   // reused as merged_b
  float* Sa = (float*)carve(NCf * 4);
  float* Sb = (float*)carve(NCf * 4);
  float* he = (float*)carve((size_t)HEN * CC * 4);
  float* lgA = (float*)carve((size_t)EE * HH * 4);
  float* lgB = (float*)carve((size_t)EE * HH * 4);
  float* stats = (float*)carve(256);    // [0:8]maxA [8:16]maxB [16:24]sumA [24:32]sumB
  unsigned short* nbfA = (unsigned short*)carve(NCf * 2);
  unsigned short* nbfB = (unsigned short*)carve(NCf * 2);

  float* out_a = (float*)d_out;
  float* out_b = out_a + NCf;

  const int cvtN  = (int)(NCf / 4);               // 5,120,000
  const int cvtNB = (cvtN + 255) / 256;           // 20000
  const int cvtW  = (CC * CC) / 4;                // 65536
  const int cvtWB = (cvtW + 255) / 256;           // 256
  dim3 gg(CC / 256, (NN + 127) / 128);            // 2 x 313
  const int heN4  = (HEN * CC) / 4;
  const int edgeT = EE * 128, edgeB = (edgeT + 255) / 256;
  const int lgT   = EE * HH,  lgB_n = (lgT + 255) / 256;

  // 1) bf16 conversions (weights transposed to [n][k] for pure-b128 staging)
  k_cvt_bf16<<<cvtNB, 256, 0, stream>>>(x_a, xbf_a, cvtN);
  k_cvt_bf16<<<cvtNB, 256, 0, stream>>>(x_b, xbf_b, cvtN);
  k_cvt_bf16_T<<<cvtWB, 256, 0, stream>>>(kW_a, wbf[0]);
  k_cvt_bf16_T<<<cvtWB, 256, 0, stream>>>(qW_a, wbf[1]);
  k_cvt_bf16_T<<<cvtWB, 256, 0, stream>>>(sW_a, wbf[2]);
  k_cvt_bf16_T<<<cvtWB, 256, 0, stream>>>(aW_a, wbf[3]);
  k_cvt_bf16_T<<<cvtWB, 256, 0, stream>>>(kW_b, wbf[4]);
  k_cvt_bf16_T<<<cvtWB, 256, 0, stream>>>(qW_b, wbf[5]);
  k_cvt_bf16_T<<<cvtWB, 256, 0, stream>>>(sW_b, wbf[6]);
  k_cvt_bf16_T<<<cvtWB, 256, 0, stream>>>(aW_b, wbf[7]);

  // 2) projection GEMMs (WMMA bf16)
  k_gemm_bf16<<<gg, 256, 0, stream>>>(xbf_a, wbf[0], kb_a, Ka, NN);
  k_gemm_bf16<<<gg, 256, 0, stream>>>(xbf_b, wbf[4], kb_b, Kb, NN);
  k_gemm_bf16<<<gg, 256, 0, stream>>>(xbf_a, wbf[1], qb_a, Qa, NN);
  k_gemm_bf16<<<gg, 256, 0, stream>>>(xbf_b, wbf[5], qb_b, Qb, NN);
  k_gemm_bf16<<<gg, 256, 0, stream>>>(xbf_a, wbf[2], sb_a, Sa, NN);
  k_gemm_bf16<<<gg, 256, 0, stream>>>(xbf_b, wbf[6], sb_b, Sb, NN);

  // 3) hyperedge accumulation
  k_zero4<<<(heN4 + 255) / 256, 256, 0, stream>>>(he, heN4);
  k_scatter_he<<<edgeB, 256, 0, stream>>>(Ka, nidx_a, hidx_a, he);
  k_scatter_he<<<edgeB, 256, 0, stream>>>(Kb, nidx_b, hidx_b, he);

  // 4) attention logits + global softmax stats
  k_init_stats<<<1, 32, 0, stream>>>(stats);
  k_logits<<<lgB_n, 256, 0, stream>>>(Qa, he, nidx_a, hidx_a, lgA, stats + 0);
  k_logits<<<lgB_n, 256, 0, stream>>>(Qb, he, nidx_b, hidx_b, lgB, stats + 8);
  k_expsum<<<lgB_n, 256, 0, stream>>>(lgA, stats + 0, stats + 16);
  k_expsum<<<lgB_n, 256, 0, stream>>>(lgB, stats + 8, stats + 24);

  // 5) weighted scatter back to nodes (K buffers reused as node_out)
  k_zero4<<<cvtNB, 256, 0, stream>>>(Ka, cvtN);
  k_zero4<<<cvtNB, 256, 0, stream>>>(Kb, cvtN);
  k_scatter_nodes<<<edgeB, 256, 0, stream>>>(he, lgA, stats + 16, nidx_a, hidx_a, Ka);
  k_scatter_nodes<<<edgeB, 256, 0, stream>>>(he, lgB, stats + 24, nidx_b, hidx_b, Kb);

  // 6) output projection GEMM + residual layernorm
  k_cvt_bf16<<<cvtNB, 256, 0, stream>>>(Ka, nbfA, cvtN);
  k_cvt_bf16<<<cvtNB, 256, 0, stream>>>(Kb, nbfB, cvtN);
  k_gemm_bf16<<<gg, 256, 0, stream>>>(nbfA, wbf[3], ab_a, Qa, NN);  // merged_a
  k_gemm_bf16<<<gg, 256, 0, stream>>>(nbfB, wbf[7], ab_b, Qb, NN);  // merged_b
  k_add_ln<<<NN, 128, 0, stream>>>(Qa, Sa, lng_a, lnb_a, out_a);
  k_add_ln<<<NN, 128, 0, stream>>>(Qb, Sb, lng_b, lnb_b, out_b);
}
